// RGCNLayer_47553877901904
// MI455X (gfx1250) — compile-verified, matrix operand order checked by reading, not run on hardware
//
#include <hip/hip_runtime.h>
#include <stdint.h>

// ---------------------------------------------------------------------------
// RGCN layer (mean aggr, 65 relations) + residual linear + BatchNorm + ReLU
// for MI455X / gfx1250.
//
// Strategy:
//   out[dst] += (x[src] @ W_rel[rel]) * (1/cnt[dst,rel])   per edge  (bf16 WMMA)
//   out[n]   += x[n] @ (W_root + W_res)                              (bf16 WMMA)
//   BatchNorm over nodes cancels per-feature constants -> bias/b_res dropped.
// Edges are counting-sorted by relation (16-aligned runs, sentinel padded) so
// every 16-edge tile shares one W_rel matrix and runs v_wmma_f32_16x16x32_bf16.
// ---------------------------------------------------------------------------

#define F       128
#define TILE_M  16
#define BLK     256   // 8 wave32s: one 16x16 output tile per wave -> 128 cols

typedef __attribute__((ext_vector_type(16))) __bf16 v16bf;
typedef __attribute__((ext_vector_type(8)))  float  v8f;

// K origin for VGPR pair v (0..7) of the 16-bit A/B operand layout
// (ISA 7.12.2: V0..3 -> K=0..7 (+8 for lanes 16..31), V4..7 -> K=16..23 (+8)).
__device__ __forceinline__ int frag_k0(int v, int hiHalf) {
  return ((v < 4) ? (2 * v) : (16 + 2 * (v - 4))) + 8 * hiHalf;
}

// ---------------------------------------------------------------------------
// small utility / prep kernels
// ---------------------------------------------------------------------------
__global__ void fill_u32(uint32_t* p, uint32_t v, long n) {
  long i = (long)blockIdx.x * blockDim.x + threadIdx.x;
  long st = (long)gridDim.x * blockDim.x;
  for (; i < n; i += st) p[i] = v;
}

__global__ void rel_hist(const int* __restrict__ et, uint32_t* cnt, long E) {
  long e = (long)blockIdx.x * blockDim.x + threadIdx.x;
  if (e < E) atomicAdd(&cnt[et[e]], 1u);
}

// serial 65-entry scan; run lengths aligned up to 16 so every tile is
// single-relation (sentinel-padded).
__global__ void rel_scan(const uint32_t* __restrict__ cnt, uint32_t* off,
                         uint32_t* cur, int R) {
  if (blockIdx.x == 0 && threadIdx.x == 0) {
    uint32_t t = 0;
    for (int r = 0; r < R; ++r) {
      off[r] = t;
      cur[r] = t;
      t += (cnt[r] + 15u) & ~15u;
    }
    off[R] = t;
  }
}

__global__ void seg_count(const int* __restrict__ ei, const int* __restrict__ et,
                          uint32_t* seg, long E, int R) {
  long e = (long)blockIdx.x * blockDim.x + threadIdx.x;
  if (e < E) {
    int dst = ei[E + e];
    int r   = et[e];
    atomicAdd(&seg[(long)dst * R + r], 1u);
  }
}

__global__ void rel_scatter(const int* __restrict__ et, uint32_t* cur,
                            uint32_t* sorted, long E) {
  long e = (long)blockIdx.x * blockDim.x + threadIdx.x;
  if (e < E) {
    uint32_t pos = atomicAdd(&cur[et[e]], 1u);
    sorted[pos] = (uint32_t)e;
  }
}

__global__ void cvt_x(const float* __restrict__ x, __bf16* xb, long n) {
  long i = (long)blockIdx.x * blockDim.x + threadIdx.x;
  if (i < n) xb[i] = (__bf16)x[i];
}

// Pack W_rel (R x F x F, row-major K x N per relation) into per-lane WMMA
// B-fragment order: [(rel*8+ntile)*4+kstep][lane][16 contiguous bf16].
__global__ void pack_w(const float* __restrict__ W, __bf16* wf, long total) {
  long i = (long)blockIdx.x * blockDim.x + threadIdx.x;
  if (i >= total) return;
  int  j    = (int)(i & 15);
  int  lane = (int)((i >> 4) & 31);
  int  ks   = (int)((i >> 9) & 3);
  int  nt   = (int)((i >> 11) & 7);
  long rel  = i >> 14;
  int hiHalf = lane >> 4, n = lane & 15, v = j >> 1, odd = j & 1;
  int k = frag_k0(v, hiHalf) + ks * 32 + odd;
  wf[i] = (__bf16)W[rel * (F * F) + (long)k * F + nt * 16 + n];
}

// Same packing for the combined root+residual weight (single 128x128).
__global__ void pack_wc(const float* __restrict__ Wroot,
                        const float* __restrict__ Wres, __bf16* wf) {
  int i = blockIdx.x * blockDim.x + threadIdx.x;
  if (i >= F * F) return;
  int j = i & 15, lane = (i >> 4) & 31, ks = (i >> 9) & 3, nt = (i >> 11) & 7;
  int hiHalf = lane >> 4, n = lane & 15, v = j >> 1, odd = j & 1;
  int k = frag_k0(v, hiHalf) + ks * 32 + odd;
  int s = k * F + nt * 16 + n;
  wf[i] = (__bf16)(Wroot[s] + Wres[s]);
}

// ---------------------------------------------------------------------------
// main edge-tile GEMM: 16 sorted edges x (128 -> 128) with one relation
// ---------------------------------------------------------------------------
__global__ void __launch_bounds__(BLK) rgcn_msg_gemm(
    const int* __restrict__ ei, const uint32_t* __restrict__ sorted,
    const uint32_t* __restrict__ off, const uint32_t* __restrict__ segcnt,
    const __bf16* __restrict__ xb, const __bf16* __restrict__ wf,
    float* __restrict__ out, long E, int R) {
  __shared__ __align__(16) __bf16 sA[TILE_M][F];
  __shared__ int   s_src[TILE_M];
  __shared__ int   s_dst[TILE_M];
  __shared__ float s_inv[TILE_M];
  __shared__ int   s_valid[TILE_M];
  __shared__ int      s_rel;
  __shared__ uint32_t s_total;

  const int tid   = threadIdx.x;
  const uint32_t base = (uint32_t)blockIdx.x * TILE_M;

  if (tid == 0) {
    uint32_t total = off[R];
    s_total = total;
    int r = 0;
    if (base < total) {
      for (int q = 0; q < R; ++q) {
        if (off[q] <= base && base < off[q + 1]) { r = q; break; }
      }
    }
    s_rel = r;
  }
  __syncthreads();
  if (base >= s_total) return;   // uniform across block

  if (tid < TILE_M) {
    uint32_t eid = sorted[base + tid];
    if (eid == 0xFFFFFFFFu) {    // sentinel padding slot
      s_src[tid] = 0; s_dst[tid] = 0; s_inv[tid] = 0.0f; s_valid[tid] = 0;
    } else {
      int src = ei[eid];
      int dst = ei[E + eid];
      uint32_t c = segcnt[(long)dst * R + s_rel];
      s_src[tid] = src; s_dst[tid] = dst;
      s_inv[tid] = 1.0f / (float)(c ? c : 1u);
      s_valid[tid] = 1;
    }
  }
  __syncthreads();

  { // stage 16x128 bf16 A tile (gathered x[src]) into LDS, 16B per thread
    int row = tid >> 4, seg = tid & 15;
    uint4 val = make_uint4(0u, 0u, 0u, 0u);
    if (s_valid[row])
      val = *(const uint4*)(xb + (long)s_src[row] * F + seg * 8);
    *(uint4*)&sA[row][seg * 8] = val;
  }
  __syncthreads();

  const int wave = tid >> 5, lane = tid & 31;
  const int hiHalf = lane >> 4, l16 = lane & 15;
  const int nt = wave;                      // output column tile 0..7
  const __bf16* wbase = wf + ((((long)s_rel * 8 + nt) * 4) * 32) * 16;

  v8f c = {};
#pragma unroll
  for (int ks = 0; ks < 4; ++ks) {          // K = 128 in 4 steps of 32
    v16bf a;
#pragma unroll
    for (int v = 0; v < 8; ++v) {
      int k0 = frag_k0(v, hiHalf) + ks * 32;
      a[2 * v]     = sA[l16][k0];
      a[2 * v + 1] = sA[l16][k0 + 1];
    }
    v16bf b = *(const v16bf*)(wbase + ((long)ks * 32 + lane) * 16);
    c = __builtin_amdgcn_wmma_f32_16x16x32_bf16(false, a, false, b,
                                                (short)0, c, false, false);
  }

  const int col = nt * 16 + l16;
#pragma unroll
  for (int v = 0; v < 8; ++v) {
    int m = v + 8 * hiHalf;                 // D row = edge slot (ISA C/D layout)
    if (s_valid[m])
      atomicAdd(&out[(long)s_dst[m] * F + col], c[v] * s_inv[m]);
  }
}

// ---------------------------------------------------------------------------
// node-tile GEMM: out[n] += x[n] @ (W_root + W_res)
// ---------------------------------------------------------------------------
__global__ void __launch_bounds__(BLK) root_gemm(const __bf16* __restrict__ xb,
                                                 const __bf16* __restrict__ wcf,
                                                 float* __restrict__ out, long N) {
  __shared__ __align__(16) __bf16 sA[TILE_M][F];
  const int tid = threadIdx.x;
  const long nbase = (long)blockIdx.x * TILE_M;
  {
    int row = tid >> 4, seg = tid & 15;
    long node = nbase + row;
    uint4 val = make_uint4(0u, 0u, 0u, 0u);
    if (node < N) val = *(const uint4*)(xb + node * F + seg * 8);
    *(uint4*)&sA[row][seg * 8] = val;
  }
  __syncthreads();

  const int wave = tid >> 5, lane = tid & 31;
  const int hiHalf = lane >> 4, l16 = lane & 15;
  const int nt = wave;
  const __bf16* wbase = wcf + ((long)nt * 4 * 32) * 16;

  v8f c = {};
#pragma unroll
  for (int ks = 0; ks < 4; ++ks) {
    v16bf a;
#pragma unroll
    for (int v = 0; v < 8; ++v) {
      int k0 = frag_k0(v, hiHalf) + ks * 32;
      a[2 * v]     = sA[l16][k0];
      a[2 * v + 1] = sA[l16][k0 + 1];
    }
    v16bf b = *(const v16bf*)(wbase + ((long)ks * 32 + lane) * 16);
    c = __builtin_amdgcn_wmma_f32_16x16x32_bf16(false, a, false, b,
                                                (short)0, c, false, false);
  }

  const int col = nt * 16 + l16;
#pragma unroll
  for (int v = 0; v < 8; ++v) {
    long node = nbase + v + 8 * hiHalf;
    if (node < N) out[node * F + col] += c[v];   // exclusive per element
  }
}

// ---------------------------------------------------------------------------
// BatchNorm (training-mode batch stats, biased var) + ReLU, in place on out
// ---------------------------------------------------------------------------
__global__ void bn_stats(const float* __restrict__ h, float* sums, float* sqs,
                         long N) {
  int f = threadIdx.x;                  // F threads
  float s = 0.0f, q = 0.0f;
  for (long r = blockIdx.x; r < N; r += gridDim.x) {
    float v = h[r * F + f];
    s += v; q += v * v;
  }
  atomicAdd(&sums[f], s);
  atomicAdd(&sqs[f], q);
}

__global__ void bn_norm(float* __restrict__ h, const float* __restrict__ sums,
                        const float* __restrict__ sqs,
                        const float* __restrict__ gamma,
                        const float* __restrict__ beta, long N) {
  long i = (long)blockIdx.x * blockDim.x + threadIdx.x;
  if (i >= N * F) return;
  int f = (int)(i & (F - 1));
  float invN = 1.0f / (float)N;
  float mean = sums[f] * invN;
  float var  = sqs[f] * invN - mean * mean;
  float v = (h[i] - mean) * rsqrtf(var + 1e-5f) * gamma[f] + beta[f];
  h[i] = v > 0.0f ? v : 0.0f;
}

// ---------------------------------------------------------------------------
extern "C" void kernel_launch(void* const* d_in, const int* in_sizes, int n_in,
                              void* d_out, int out_size, void* d_ws, size_t ws_size,
                              hipStream_t stream) {
  const float* x     = (const float*)d_in[0];
  const int*   ei    = (const int*)  d_in[1];   // (2,E): src row then dst row
  const int*   et    = (const int*)  d_in[2];   // (E,)
  const float* Wrel  = (const float*)d_in[3];   // (R,F,F)
  const float* Wroot = (const float*)d_in[4];   // (F,F)
  /* d_in[5] bias, d_in[7] b_res: per-feature constants -> cancelled by BN */
  const float* Wres  = (const float*)d_in[6];   // (F,F)
  const float* gamma = (const float*)d_in[8];
  const float* beta  = (const float*)d_in[9];
  float* out = (float*)d_out;

  const long N = in_sizes[0] / F;
  const long E = in_sizes[2];
  const long R = in_sizes[3] / (F * F);
  (void)n_in; (void)out_size; (void)ws_size;

  // ---- carve workspace (all sections 256B aligned; ~15.5 MB total) ----
  char* ws = (char*)d_ws;
  size_t o = 0;
  auto carve = [&](size_t bytes) -> char* {
    size_t a = (o + 255) & ~(size_t)255;
    o = a + bytes;
    return ws + a;
  };
  uint32_t* relcnt = (uint32_t*)carve((size_t)R * 4);
  uint32_t* off    = (uint32_t*)carve((size_t)(R + 1) * 4);
  uint32_t* cur    = (uint32_t*)carve((size_t)R * 4);
  uint32_t* segcnt = (uint32_t*)carve((size_t)N * R * 4);
  const long cap = E + 16 * R;                     // sentinel-padded capacity
  uint32_t* sorted = (uint32_t*)carve((size_t)cap * 4);
  __bf16* xb  = (__bf16*)carve((size_t)N * F * 2);
  __bf16* wf  = (__bf16*)carve((size_t)R * F * F * 2);
  __bf16* wcf = (__bf16*)carve((size_t)F * F * 2);
  float* sums = (float*)carve((size_t)F * 4);
  float* sqs  = (float*)carve((size_t)F * 4);

  auto nb = [](long n, int b) { return (unsigned)((n + b - 1) / b); };

  // ---- init (every call: harness does not re-clear buffers) ----
  fill_u32<<<nb(R, 256), 256, 0, stream>>>(relcnt, 0u, R);
  fill_u32<<<nb(N * R, 256), 256, 0, stream>>>(segcnt, 0u, N * R);
  fill_u32<<<nb(cap, 256), 256, 0, stream>>>(sorted, 0xFFFFFFFFu, cap);
  fill_u32<<<1, 256, 0, stream>>>((uint32_t*)sums, 0u, F);
  fill_u32<<<1, 256, 0, stream>>>((uint32_t*)sqs, 0u, F);
  fill_u32<<<nb(N * F, 256), 256, 0, stream>>>((uint32_t*)out, 0u, N * F);

  // ---- counting sort of edges by relation (16-aligned runs) ----
  rel_hist<<<nb(E, 256), 256, 0, stream>>>(et, relcnt, E);
  rel_scan<<<1, 1, 0, stream>>>(relcnt, off, cur, (int)R);
  seg_count<<<nb(E, 256), 256, 0, stream>>>(ei, et, segcnt, E, (int)R);
  rel_scatter<<<nb(E, 256), 256, 0, stream>>>(et, cur, sorted, E);

  // ---- bf16 conversion + WMMA fragment packing ----
  cvt_x<<<nb(N * F, 256), 256, 0, stream>>>(x, xb, N * F);
  pack_w<<<nb(R * F * F, 256), 256, 0, stream>>>(Wrel, wf, R * F * F);
  pack_wc<<<nb(F * F, 256), 256, 0, stream>>>(Wroot, Wres, wcf);

  // ---- main compute ----
  const long ntiles = (E + TILE_M - 1) / TILE_M + R;  // >= padded_total/16
  rgcn_msg_gemm<<<(unsigned)ntiles, BLK, 0, stream>>>(ei, sorted, off, segcnt,
                                                      xb, wf, out, E, (int)R);
  root_gemm<<<nb(N, TILE_M), BLK, 0, stream>>>(xb, wcf, out, N);

  // ---- BatchNorm + ReLU ----
  bn_stats<<<256, F, 0, stream>>>(out, sums, sqs, N);
  bn_norm<<<nb(N * F, 256), 256, 0, stream>>>(out, sums, sqs, gamma, beta, N);
}
